// ConvLSTM_res_18287970746517
// MI455X (gfx1250) — matrix operand order repaired and checked by reading, not compiled, not run on hardware
//
#include <hip/hip_runtime.h>

// ---------------------------------------------------------------------------
// ConvLSTM (N=512, CIN=128, CH=256, W=8) as implicit GEMM on CDNA5 WMMA.
//   gates[n][1024][64] = Wcat[1024][K] * patches[n][K][64],  K = 9*128 + 9*256
// K-order: (kh,kw)-major, channel-minor  ->  one (kh,kw) tap per 32-wide K tile.
// bf16 A/B fragments, f32 accumulation via v_wmma_f32_16x16x32_bf16.
// ---------------------------------------------------------------------------

typedef __attribute__((ext_vector_type(16))) __bf16 bf16x16;
typedef __attribute__((ext_vector_type(8)))  float  v8f;

#define NB    512
#define CIN   128
#define CH    256
#define SP    64          // 8x8 spatial
#define MTOT  1024        // 4 gates * 256 ch
#define KTX   36          // 9 taps * (128/32) k-tiles (x part)
#define KTH   72          // 9 taps * (256/32) k-tiles (h part)
#define WPX_ELEMS ((size_t)64 * KTX * 32 * 16)   // [mtile][ktile][lane][16 bf16]
#define WPH_ELEMS ((size_t)64 * KTH * 32 * 16)
#define LDS_BYTES 262144                         // max(pads 76800, gates 262144)

__device__ __forceinline__ unsigned short f2bf(float f) {
    union { float f; unsigned int u; } v; v.f = f;
    unsigned int u = v.u;
    u += 0x7FFFu + ((u >> 16) & 1u);             // round-to-nearest-even
    return (unsigned short)(u >> 16);
}
__device__ __forceinline__ float fsigmoid(float x) {
    return 1.0f / (1.0f + __expf(-x));
}
__device__ __forceinline__ float ftanh(float x) {
    float e = __expf(2.0f * x);
    return (e - 1.0f) / (e + 1.0f);
}
// A-matrix 16x32 lane layout (contiguous, matches B): K for (half, element i) = half*16 + i
__device__ __forceinline__ int a_klocal(int half, int i) {
    return half * 16 + i;
}

// ---------------------------------------------------------------------------
// Pack kernel: permute f32 OIHW weights into per-lane bf16 A-fragment order,
// and precompute bias sums (b_i + b_h).
// ---------------------------------------------------------------------------
__global__ void convlstm_pack(
    const float* w_ii, const float* w_if, const float* w_ig, const float* w_io,
    const float* w_hi, const float* w_hf, const float* w_hg, const float* w_ho,
    const float* b_ii, const float* b_if, const float* b_ig, const float* b_io,
    const float* b_hi, const float* b_hf, const float* b_hg, const float* b_ho,
    unsigned short* wpx, unsigned short* wph, float* bsum)
{
    const int NX = 64 * KTX * 32;
    const int NH = 64 * KTH * 32;
    int idx = blockIdx.x * blockDim.x + threadIdx.x;

    if (idx < NX) {                                   // x-side weights, C=128
        int lane = idx & 31, t = idx >> 5;
        int kt = t % KTX, mt = t / KTX;
        int m = mt * 16 + (lane & 15), hl = lane >> 4;
        int gate = m >> 8, ch = m & 255;
        const float* Wg = gate == 0 ? w_ii : gate == 1 ? w_if : gate == 2 ? w_ig : w_io;
        unsigned short* dst = wpx + ((size_t)(mt * KTX + kt) * 32 + lane) * 16;
#pragma unroll
        for (int i = 0; i < 16; ++i) {
            int k = kt * 32 + a_klocal(hl, i);
            int off = k >> 7, c = k & 127;            // k = off*128 + c
            dst[i] = f2bf(Wg[((size_t)ch * CIN + c) * 9 + off]);
        }
    } else if (idx < NX + NH) {                       // h-side weights, C=256
        int e = idx - NX;
        int lane = e & 31, t = e >> 5;
        int kt = t % KTH, mt = t / KTH;
        int m = mt * 16 + (lane & 15), hl = lane >> 4;
        int gate = m >> 8, ch = m & 255;
        const float* Wg = gate == 0 ? w_hi : gate == 1 ? w_hf : gate == 2 ? w_hg : w_ho;
        unsigned short* dst = wph + ((size_t)(mt * KTH + kt) * 32 + lane) * 16;
#pragma unroll
        for (int i = 0; i < 16; ++i) {
            int k = kt * 32 + a_klocal(hl, i);
            int off = k >> 8, c = k & 255;            // k = off*256 + c
            dst[i] = f2bf(Wg[((size_t)ch * CH + c) * 9 + off]);
        }
    } else if (idx < NX + NH + MTOT) {                // bias sums
        int m = idx - NX - NH;
        int gate = m >> 8, ch = m & 255;
        const float* bi = gate == 0 ? b_ii : gate == 1 ? b_if : gate == 2 ? b_ig : b_io;
        const float* bh = gate == 0 ? b_hi : gate == 1 ? b_hf : gate == 2 ? b_hg : b_ho;
        bsum[m] = bi[ch] + bh[ch];
    }
}

// ---------------------------------------------------------------------------
// Main kernel: one workgroup (16 waves) per batch element.
//   Phase A: reflect-pad x/h0 into channel-last bf16 LDS images.
//   Phase B: 108 K-tiles of 16x16x32 bf16 WMMA per wave (1728 wmma/wave).
//   Phase C: stage f32 gates[1024][64] in (reused) LDS.
//   Phase D: LSTM pointwise epilogue -> o, h_t, c_t.
// ---------------------------------------------------------------------------
__global__ __launch_bounds__(512, 1)
void convlstm_main(const float* __restrict__ x, const float* __restrict__ hid,
                   const unsigned short* __restrict__ wpx,
                   const unsigned short* __restrict__ wph,
                   const float* __restrict__ bsum,
                   const float* __restrict__ wc_i, const float* __restrict__ wc_f,
                   const float* __restrict__ wc_o, float* __restrict__ out)
{
    extern __shared__ char smem[];
    unsigned short* hpadT = (unsigned short*)smem;            // [10][10][256] bf16
    unsigned short* xpadT = (unsigned short*)(smem + 51200);  // [10][10][128] bf16

    const int n   = blockIdx.x;
    const float* xg = x   + (size_t)n * CIN * SP;
    const float* h0 = hid + (size_t)n * 2 * CH * SP;
    const float* c0 = h0 + CH * SP;

    // ---- Phase A: build reflect-padded channel-last bf16 images in LDS ----
    for (int e = threadIdx.x; e < 100 * CH + 100 * CIN; e += blockDim.x) {
        int pos, c; const float* src; unsigned short* dst;
        if (e < 100 * CH) { pos = e >> 8; c = e & 255; src = h0 + c * SP; dst = hpadT + pos * CH + c; }
        else { int e2 = e - 100 * CH; pos = e2 >> 7; c = e2 & 127; src = xg + c * SP; dst = xpadT + pos * CIN + c; }
        int py = pos / 10, px = pos % 10;
        int sy = py - 1; sy = sy < 0 ? 1 : (sy > 7 ? 6 : sy);   // reflect pad=1
        int sx = px - 1; sx = sx < 0 ? 1 : (sx > 7 ? 6 : sx);
        *dst = f2bf(src[sy * 8 + sx]);
    }
    __syncthreads();

    // ---- Phase B: implicit-GEMM via bf16 WMMA ----
    const int wave = threadIdx.x >> 5;
    const int lane = threadIdx.x & 31;
    const int hl   = lane >> 4;        // lane half (K/N split per ISA layout)
    const int col  = lane & 15;

    int by[4], bx[4];
#pragma unroll
    for (int t = 0; t < 4; ++t) { int np = t * 16 + col; by[t] = np >> 3; bx[t] = np & 7; }

    v8f acc[4][4];
#pragma unroll
    for (int a = 0; a < 4; ++a)
#pragma unroll
        for (int t = 0; t < 4; ++t) acc[a][t] = (v8f){0,0,0,0,0,0,0,0};

    // x part: 36 K-tiles (C=128 -> 4 tiles per tap)
    for (int kt = 0; kt < KTX; ++kt) {
        int off = kt >> 2, kh = off / 3, kw = off % 3;
        int cb = (kt & 3) * 32 + hl * 16;
        bf16x16 afr[4];
#pragma unroll
        for (int a = 0; a < 4; ++a) {
            int mt = wave * 4 + a;
            afr[a] = *(const bf16x16*)(wpx + ((size_t)(mt * KTX + kt) * 32 + lane) * 16);
        }
#pragma unroll
        for (int t = 0; t < 4; ++t) {
            bf16x16 bfr = *(const bf16x16*)(xpadT + ((by[t] + kh) * 10 + bx[t] + kw) * CIN + cb);
#pragma unroll
            for (int a = 0; a < 4; ++a)
                acc[a][t] = __builtin_amdgcn_wmma_f32_16x16x32_bf16(
                    false, afr[a], false, bfr, (short)0, acc[a][t], false, false);
        }
    }
    // h part: 72 K-tiles (C=256 -> 8 tiles per tap)
    for (int kt = 0; kt < KTH; ++kt) {
        int off = kt >> 3, kh = off / 3, kw = off % 3;
        int cb = (kt & 7) * 32 + hl * 16;
        bf16x16 afr[4];
#pragma unroll
        for (int a = 0; a < 4; ++a) {
            int mt = wave * 4 + a;
            afr[a] = *(const bf16x16*)(wph + ((size_t)(mt * KTH + kt) * 32 + lane) * 16);
        }
#pragma unroll
        for (int t = 0; t < 4; ++t) {
            bf16x16 bfr = *(const bf16x16*)(hpadT + ((by[t] + kh) * 10 + bx[t] + kw) * CH + cb);
#pragma unroll
            for (int a = 0; a < 4; ++a)
                acc[a][t] = __builtin_amdgcn_wmma_f32_16x16x32_bf16(
                    false, afr[a], false, bfr, (short)0, acc[a][t], false, false);
        }
    }
    __syncthreads();   // all waves done reading padded images

    // ---- Phase C: stage gates[1024][64] f32 in reused LDS ----
    float* gl = (float*)smem;
#pragma unroll
    for (int a = 0; a < 4; ++a) {
        int mbase = (wave * 4 + a) * 16 + hl * 8;
#pragma unroll
        for (int t = 0; t < 4; ++t) {
            int np = t * 16 + col;
#pragma unroll
            for (int r = 0; r < 8; ++r) {            // C/D layout: VGPR r -> M = hl*8 + r
                int m = mbase + r;
                gl[m * SP + np] = acc[a][t][r] + bsum[m];
            }
        }
    }
    __syncthreads();

    // ---- Phase D: LSTM pointwise epilogue ----
    float* out_o = out;                                       // [512][256][64]
    float* out_h = out + (size_t)NB * CH * SP;                // [512][2][256][64]
    for (int e = threadIdx.x; e < CH * SP; e += blockDim.x) {
        int ch = e >> 6, np = e & 63;
        float gi = gl[(0 * CH + ch) * SP + np];
        float gf = gl[(1 * CH + ch) * SP + np];
        float gg = gl[(2 * CH + ch) * SP + np];
        float go = gl[(3 * CH + ch) * SP + np];
        float c0v = c0[e], h0v = h0[e];
        float iv = fsigmoid(gi + c0v * wc_i[e]);
        float fv = fsigmoid(gf + c0v * wc_f[e]);
        float gv = ftanh(gg);
        float ct = c0v + fv * c0v + iv * gv;
        float ov = fsigmoid(go + ct * wc_o[e]);
        float ht = h0v + ov * ftanh(ct);
        out_o[(size_t)n * CH * SP + e] = ov;
        out_h[(size_t)n * 2 * CH * SP + e] = ht;              // h_t  (slot 0)
        out_h[(size_t)n * 2 * CH * SP + CH * SP + e] = ct;    // c_t  (slot 1)
    }
}

// ---------------------------------------------------------------------------
extern "C" void kernel_launch(void* const* d_in, const int* in_sizes, int n_in,
                              void* d_out, int out_size, void* d_ws, size_t ws_size,
                              hipStream_t stream) {
    const float* x    = (const float*)d_in[0];
    const float* hid  = (const float*)d_in[1];
    const float* w_ii = (const float*)d_in[2];  const float* b_ii = (const float*)d_in[3];
    const float* w_if = (const float*)d_in[4];  const float* b_if = (const float*)d_in[5];
    const float* w_ig = (const float*)d_in[6];  const float* b_ig = (const float*)d_in[7];
    const float* w_io = (const float*)d_in[8];  const float* b_io = (const float*)d_in[9];
    const float* w_hi = (const float*)d_in[10]; const float* b_hi = (const float*)d_in[11];
    const float* w_hf = (const float*)d_in[12]; const float* b_hf = (const float*)d_in[13];
    const float* w_hg = (const float*)d_in[14]; const float* b_hg = (const float*)d_in[15];
    const float* w_ho = (const float*)d_in[16]; const float* b_ho = (const float*)d_in[17];
    const float* wc_i = (const float*)d_in[18];
    const float* wc_f = (const float*)d_in[19];
    const float* wc_o = (const float*)d_in[20];

    // workspace: packed bf16 weights (~7.1 MB) + bias sums
    unsigned short* wpx = (unsigned short*)d_ws;
    unsigned short* wph = wpx + WPX_ELEMS;
    float* bsum = (float*)(wph + WPH_ELEMS);

    const int packN = 64 * KTX * 32 + 64 * KTH * 32 + MTOT;
    convlstm_pack<<<(packN + 255) / 256, 256, 0, stream>>>(
        w_ii, w_if, w_ig, w_io, w_hi, w_hf, w_hg, w_ho,
        b_ii, b_if, b_ig, b_io, b_hi, b_hf, b_hg, b_ho,
        wpx, wph, bsum);

    convlstm_main<<<NB, 512, LDS_BYTES, stream>>>(
        x, hid, wpx, wph, bsum, wc_i, wc_f, wc_o, (float*)d_out);
}
